// AttnBlock_45286135169066
// MI455X (gfx1250) — matrix operand order repaired and checked by reading, not compiled, not run on hardware
//
#include <hip/hip_runtime.h>

#define CCH 256
#define NVOX 4096
#define NB 2

typedef __bf16 bf16_t;
typedef bf16_t v16bf __attribute__((ext_vector_type(16)));
typedef bf16_t v8bf  __attribute__((ext_vector_type(8)));
typedef float  v8f   __attribute__((ext_vector_type(8)));
typedef unsigned int u32x4 __attribute__((ext_vector_type(4)));
typedef int i32x4 __attribute__((ext_vector_type(4)));
typedef int i32x8 __attribute__((ext_vector_type(8)));

// Native fptrunc: backend selects the gfx1250 bf16 convert (RNE).
static __device__ inline bf16_t f2bf(float f) { return (bf16_t)f; }

static __device__ inline v8f wmma_bf16(v16bf a, v16bf b, v8f c) {
  return __builtin_amdgcn_wmma_f32_16x16x32_bf16(false, a, false, b, (short)0, c,
                                                 false, false);
}

// B-fragment: 16 contiguous bf16 K-values for this lane (two b128 loads)
static __device__ inline v16bf ldB(const bf16_t* p) {
  v8bf lo = *(const v8bf*)p;
  v8bf hi = *(const v8bf*)(p + 8);
  return __builtin_shufflevector(lo, hi, 0,1,2,3,4,5,6,7,8,9,10,11,12,13,14,15);
}
// A-fragment: two 8-element bf16 K-chunks; every caller passes adjacent ones (16 contiguous K)
static __device__ inline v16bf ldA2(const bf16_t* p0, const bf16_t* p1) {
  v8bf lo = *(const v8bf*)p0;
  v8bf hi = *(const v8bf*)p1;
  return __builtin_shufflevector(lo, hi, 0,1,2,3,4,5,6,7,8,9,10,11,12,13,14,15);
}

static __device__ inline float redmax16(float v) {
#pragma unroll
  for (int m = 1; m < 16; m <<= 1) v = fmaxf(v, __shfl_xor(v, m, 32));
  return v;
}
static __device__ inline float redsum16(float v) {
#pragma unroll
  for (int m = 1; m < 16; m <<= 1) v += __shfl_xor(v, m, 32);
  return v;
}

// Wave-local LDS drain (wave-private buffers: no block barrier needed)
static __device__ inline void wait_ds0() {
  __asm__ volatile("s_wait_dscnt 0x0" ::: "memory");
}

// TDM: issue a 2D tensor_load_to_lds.  g1 carries dims/strides, group0 built
// here from addresses.  D# layout per CDNA5 ISA section 8.3/8.4.
static __device__ inline void tdm_load2d(unsigned lds_addr,
                                         unsigned long long gaddr, i32x8 g1) {
  u32x4 g0;
  g0[0] = 1u;                                              // count=1, user desc
  g0[1] = lds_addr;                                        // lds_addr [63:32]
  g0[2] = (unsigned)gaddr;                                 // global_addr lo
  g0[3] = (unsigned)((gaddr >> 32) & 0x01ffffffull) | (2u << 30);  // hi | type=2
  i32x4 z4; z4[0] = 0; z4[1] = 0; z4[2] = 0; z4[3] = 0;
#if __clang_major__ >= 23
  i32x8 z8;
#pragma unroll
  for (int i = 0; i < 8; ++i) z8[i] = 0;
  __builtin_amdgcn_tensor_load_to_lds(g0, g1, z4, z4, z8, 0);
#else
  __builtin_amdgcn_tensor_load_to_lds(g0, g1, z4, z4, 0);
#endif
}

// ---------------------------------------------------------------------------
// 0) Convert the four 256x256 fp32 weight matrices to bf16 once.
// ---------------------------------------------------------------------------
__global__ __launch_bounds__(256) void wprep_kernel(const float* __restrict__ w0,
                                                    const float* __restrict__ w1,
                                                    const float* __restrict__ w2,
                                                    const float* __restrict__ w3,
                                                    bf16_t* __restrict__ wb) {
  int i = blockIdx.x * 256 + threadIdx.x;  // 65536 total
  wb[i]              = f2bf(w0[i]);
  wb[65536 + i]      = f2bf(w1[i]);
  wb[2 * 65536 + i]  = f2bf(w2[i]);
  wb[3 * 65536 + i]  = f2bf(w3[i]);
}

// ---------------------------------------------------------------------------
// 1) GroupNorm: one block per (batch, group). Writes hT [B, N, C] bf16.
// ---------------------------------------------------------------------------
__global__ __launch_bounds__(256) void gn_kernel(const float* __restrict__ x,
                                                 const float* __restrict__ gamma,
                                                 const float* __restrict__ beta,
                                                 bf16_t* __restrict__ hT) {
  int b = blockIdx.x >> 5;
  int g = blockIdx.x & 31;
  const float* xb = x + ((size_t)b * CCH + g * 8) * NVOX;
  int tid = threadIdx.x;

  float s = 0.f, ss = 0.f;
  for (int c = 0; c < 8; ++c)
    for (int n = tid; n < NVOX; n += 256) {
      float v = xb[c * NVOX + n];
      s += v; ss += v * v;
    }
  __shared__ float rs[256], rq[256];
  rs[tid] = s; rq[tid] = ss;
  __syncthreads();
  for (int o = 128; o > 0; o >>= 1) {
    if (tid < o) { rs[tid] += rs[tid + o]; rq[tid] += rq[tid + o]; }
    __syncthreads();
  }
  const float invn = 1.0f / (8.0f * NVOX);
  float mu  = rs[0] * invn;
  float var = rq[0] * invn - mu * mu;
  float inv = rsqrtf(var + 1e-6f);

  float gm[8], bt[8];
#pragma unroll
  for (int c = 0; c < 8; ++c) { gm[c] = gamma[g * 8 + c]; bt[c] = beta[g * 8 + c]; }

  for (int n = tid; n < NVOX; n += 256) {
    v8bf o8;
#pragma unroll
    for (int c = 0; c < 8; ++c) {
      float v = (xb[c * NVOX + n] - mu) * inv * gm[c] + bt[c];
      o8[c] = f2bf(v);
    }
    *(v8bf*)&hT[((size_t)b * NVOX + n) * CCH + g * 8] = o8;  // 16B store
  }
}

// ---------------------------------------------------------------------------
// 2) q/k/v = w{0,1,2} @ h + b.  One 16x16 tile per wave, K=256 (8 WMMAs each).
//    q,k stored transposed [B,N,C]; v stored [B,C,N]; all bf16. scale folded
//    into q.
// ---------------------------------------------------------------------------
__global__ __launch_bounds__(128) void qkv_kernel(
    const bf16_t* __restrict__ hT, const bf16_t* __restrict__ wb,
    const float* __restrict__ b0, const float* __restrict__ b1,
    const float* __restrict__ b2,
    bf16_t* __restrict__ qT, bf16_t* __restrict__ kT, bf16_t* __restrict__ vB) {
  __shared__ __attribute__((aligned(16))) bf16_t lds_t[4][16 * 16];

  int tid  = threadIdx.x;
  int wv   = tid >> 5;
  int lane = tid & 31;
  int lm   = lane & 15;
  int half = lane >> 4;

  int tile = blockIdx.x * 4 + wv;        // 8192 tiles total
  int b  = tile >> 12;
  int mt = ((tile >> 8) & 15) << 4;      // output-channel tile
  int nt = (tile & 255) << 4;            // voxel tile

  // B fragments of h (shared by q,k,v GEMMs)
  v16bf hfrag[8];
  const bf16_t* hrow = hT + ((size_t)b * NVOX + (nt + lm)) * CCH;
#pragma unroll
  for (int kk = 0; kk < 8; ++kk)
    hfrag[kk] = ldB(hrow + kk * 32 + half * 16);

  const float* bs3[3] = {b0, b1, b2};

#pragma unroll
  for (int mi = 0; mi < 3; ++mi) {
    v8f acc;
#pragma unroll
    for (int r = 0; r < 8; ++r) acc[r] = 0.f;
    const bf16_t* wrow = wb + (size_t)mi * 65536 + (size_t)(mt + lm) * CCH;
#pragma unroll
    for (int kk = 0; kk < 8; ++kk) {
      int c0 = kk * 32 + 16 * half;
      acc = wmma_bf16(ldA2(wrow + c0, wrow + c0 + 8), hfrag[kk], acc);
    }
#pragma unroll
    for (int r = 0; r < 8; ++r) acc[r] += bs3[mi][mt + r + 8 * half];
    if (mi == 0) {
#pragma unroll
      for (int r = 0; r < 8; ++r) acc[r] *= 0.0625f;  // 256^-0.5
    }
    if (mi < 2) {
      // transpose through wave-private LDS, write qT/kT with 16B stores
#pragma unroll
      for (int r = 0; r < 8; ++r)
        lds_t[wv][(r + 8 * half) * 16 + lm] = f2bf(acc[r]);
      wait_ds0();
      bf16_t* dst = (mi == 0) ? qT : kT;
      int row = lane >> 1, hc = lane & 1;
      v8bf rv = *(v8bf*)&lds_t[wv][row * 16 + hc * 8];
      *(v8bf*)&dst[((size_t)b * NVOX + nt + row) * CCH + mt + hc * 8] = rv;
      wait_ds0();
    } else {
      // v stays channel-major [B,C,N]
#pragma unroll
      for (int r = 0; r < 8; ++r)
        vB[((size_t)b * CCH + mt + r + 8 * half) * NVOX + nt + lm] = f2bf(acc[r]);
    }
  }
}

// ---------------------------------------------------------------------------
// 3) Flash attention. Block = 4 waves x 16 query rows; 128 key tiles of 32.
//    k/v tiles double-buffered in LDS, staged by the Tensor Data Mover with
//    one-tile lookahead (DMA overlaps WMMA+softmax).  P converted D->A layout
//    via wave-private LDS buffer.  Writes oT [B,N,C] bf16.
// ---------------------------------------------------------------------------
__global__ __launch_bounds__(128, 1) void attn_kernel(
    const bf16_t* __restrict__ qT, const bf16_t* __restrict__ kT,
    const bf16_t* __restrict__ vB, bf16_t* __restrict__ oT) {
  __shared__ __attribute__((aligned(16))) bf16_t lds_k[2][32 * 256];
  __shared__ __attribute__((aligned(16))) bf16_t lds_v[2][32 * 256];
  __shared__ __attribute__((aligned(16))) bf16_t lds_p[4][16 * 32];

  int tid  = threadIdx.x;
  int wv   = tid >> 5;
  int lane = tid & 31;
  int lm   = lane & 15;
  int half = lane >> 4;

  int b  = blockIdx.x >> 6;
  int q0 = (blockIdx.x & 63) * 64 + wv * 16;

  // Q A-fragments (resident in VGPRs across the whole key loop)
  v16bf qA[8];
  const bf16_t* qrow = qT + ((size_t)b * NVOX + (q0 + lm)) * CCH;
#pragma unroll
  for (int kk = 0; kk < 8; ++kk) {
    int c0 = kk * 32 + 16 * half;
    qA[kk] = ldA2(qrow + c0, qrow + c0 + 8);
  }

  v8f o[16];
#pragma unroll
  for (int ct = 0; ct < 16; ++ct)
#pragma unroll
    for (int r = 0; r < 8; ++r) o[ct][r] = 0.f;
  float mrow[8], lrow[8];
#pragma unroll
  for (int r = 0; r < 8; ++r) { mrow[r] = -3.0e38f; lrow[r] = 0.f; }

  // TDM descriptors (group1): dims/strides per D# bit layout.
  // kT tile: tile 256x32, tensor_dim0=256 (stride 256), tensor_dim1=4096.
  i32x8 g1k;
  g1k[0] = 0x00010000;  // data_size=2B
  g1k[1] = 0x01000000;  // tensor_dim0=256 -> bits[63:48]
  g1k[2] = 0x10000000;  // tensor_dim1=4096 -> bits[95:80]
  g1k[3] = 0x01000000;  // tile_dim0=256 -> bits[127:112]
  g1k[4] = 32;          // tile_dim1=32
  g1k[5] = 256;         // tensor_dim0_stride
  g1k[6] = 0; g1k[7] = 0;
  // v tile: tile 32x256, tensor_dim0=4096 (stride 4096), tensor_dim1=256.
  i32x8 g1v;
  g1v[0] = 0x00010000;
  g1v[1] = 0x10000000;  // tensor_dim0=4096
  g1v[2] = 0x01000000;  // tensor_dim1=256
  g1v[3] = 0x00200000;  // tile_dim0=32
  g1v[4] = 256;         // tile_dim1=256
  g1v[5] = 4096;        // tensor_dim0_stride
  g1v[6] = 0; g1v[7] = 0;

  const bf16_t* kbat = kT + (size_t)b * NVOX * CCH;
  const bf16_t* vbat = vB + (size_t)b * CCH * NVOX;
  unsigned lkb[2], lvb[2];
  lkb[0] = (unsigned)(unsigned long long)(void*)lds_k[0];
  lkb[1] = (unsigned)(unsigned long long)(void*)lds_k[1];
  lvb[0] = (unsigned)(unsigned long long)(void*)lds_v[0];
  lvb[1] = (unsigned)(unsigned long long)(void*)lds_v[1];

  // Prologue: kick off tile 0.
  if (wv == 0) {
    tdm_load2d(lkb[0], (unsigned long long)(const void*)kbat, g1k);
    tdm_load2d(lvb[0], (unsigned long long)(const void*)vbat, g1v);
  }

  for (int kt = 0; kt < 128; ++kt) {
    int cur = kt & 1;
    if (wv == 0) {
      if (kt + 1 < 128) {
        int m1 = (kt + 1) * 32;
        tdm_load2d(lkb[cur ^ 1],
                   (unsigned long long)(const void*)(kbat + (size_t)m1 * CCH),
                   g1k);
        tdm_load2d(lvb[cur ^ 1],
                   (unsigned long long)(const void*)(vbat + m1), g1v);
        // TDM completes in order: <=2 outstanding means tile kt has landed.
        __builtin_amdgcn_s_wait_tensorcnt(2);
      } else {
        __builtin_amdgcn_s_wait_tensorcnt(0);
      }
    }
    __syncthreads();
    const bf16_t* Kc = lds_k[cur];
    const bf16_t* Vc = lds_v[cur];

    // S = Q K^T (two 16x16 key sub-tiles)
    v8f s0, s1;
#pragma unroll
    for (int r = 0; r < 8; ++r) { s0[r] = 0.f; s1[r] = 0.f; }
#pragma unroll
    for (int kk = 0; kk < 8; ++kk) {
      int c = kk * 32 + 16 * half;
      s0 = wmma_bf16(qA[kk], ldB(&Kc[(size_t)lm * 256 + c]), s0);
      s1 = wmma_bf16(qA[kk], ldB(&Kc[(size_t)(16 + lm) * 256 + c]), s1);
    }

    // online softmax update
    float p0[8], p1[8];
#pragma unroll
    for (int r = 0; r < 8; ++r) {
      float vm = redmax16(fmaxf(s0[r], s1[r]));
      float mnew = fmaxf(mrow[r], vm);
      float alpha = __expf(mrow[r] - mnew);
      p0[r] = __expf(s0[r] - mnew);
      p1[r] = __expf(s1[r] - mnew);
      float rsum = redsum16(p0[r] + p1[r]);
      lrow[r] = lrow[r] * alpha + rsum;
      mrow[r] = mnew;
#pragma unroll
      for (int ct = 0; ct < 16; ++ct) o[ct][r] *= alpha;
    }

    // P: D-layout -> A-layout via wave-private LDS buffer
#pragma unroll
    for (int r = 0; r < 8; ++r) {
      int row = r + 8 * half;
      lds_p[wv][row * 32 + lm]      = f2bf(p0[r]);
      lds_p[wv][row * 32 + 16 + lm] = f2bf(p1[r]);
    }
    wait_ds0();
    v16bf pA = ldA2(&lds_p[wv][lm * 32 + 16 * half],
                    &lds_p[wv][lm * 32 + 8 + 16 * half]);

    // O += P V   (16 channel tiles, K = 32 keys)
#pragma unroll
    for (int ct = 0; ct < 16; ++ct)
      o[ct] = wmma_bf16(pA, ldB(&Vc[(size_t)(ct * 16 + lm) * 32 + 16 * half]),
                        o[ct]);
    __syncthreads();  // all waves done with buf[cur] before it is re-filled
  }

  // normalize + store oT [B,N,C] bf16 (coalesced 2B x 16 lanes)
#pragma unroll
  for (int r = 0; r < 8; ++r) {
    float il = 1.0f / lrow[r];
    size_t rowbase = ((size_t)b * NVOX + q0 + r + 8 * half) * CCH;
#pragma unroll
    for (int ct = 0; ct < 16; ++ct)
      oT[rowbase + ct * 16 + lm] = f2bf(o[ct][r] * il);
  }
}

// ---------------------------------------------------------------------------
// 4) y = x + w3 @ out + b3
// ---------------------------------------------------------------------------
__global__ __launch_bounds__(128) void out_kernel(const bf16_t* __restrict__ oT,
                                                  const bf16_t* __restrict__ wb3,
                                                  const float* __restrict__ b3,
                                                  const float* __restrict__ x,
                                                  float* __restrict__ y) {
  int tid  = threadIdx.x;
  int wv   = tid >> 5;
  int lane = tid & 31;
  int lm   = lane & 15;
  int half = lane >> 4;

  int tile = blockIdx.x * 4 + wv;
  int b  = tile >> 12;
  int mt = ((tile >> 8) & 15) << 4;
  int nt = (tile & 255) << 4;

  const bf16_t* orow = oT + ((size_t)b * NVOX + (nt + lm)) * CCH;
  const bf16_t* wrow = wb3 + (size_t)(mt + lm) * CCH;

  v8f acc;
#pragma unroll
  for (int r = 0; r < 8; ++r) acc[r] = 0.f;
#pragma unroll
  for (int kk = 0; kk < 8; ++kk) {
    int c0 = kk * 32 + 16 * half;
    acc = wmma_bf16(ldA2(wrow + c0, wrow + c0 + 8),
                    ldB(orow + kk * 32 + half * 16), acc);
  }

#pragma unroll
  for (int r = 0; r < 8; ++r) {
    int c = mt + r + 8 * half;
    size_t idx = ((size_t)b * CCH + c) * NVOX + nt + lm;
    y[idx] = x[idx] + acc[r] + b3[c];
  }
}

// ---------------------------------------------------------------------------
extern "C" void kernel_launch(void* const* d_in, const int* in_sizes, int n_in,
                              void* d_out, int out_size, void* d_ws, size_t ws_size,
                              hipStream_t stream) {
  (void)in_sizes; (void)n_in; (void)out_size; (void)ws_size;
  const float* x     = (const float*)d_in[0];
  const float* gamma = (const float*)d_in[1];
  const float* beta  = (const float*)d_in[2];
  const float* w0 = (const float*)d_in[3];
  const float* b0 = (const float*)d_in[4];
  const float* w1 = (const float*)d_in[5];
  const float* b1 = (const float*)d_in[6];
  const float* w2 = (const float*)d_in[7];
  const float* b2 = (const float*)d_in[8];
  const float* w3 = (const float*)d_in[9];
  const float* b3 = (const float*)d_in[10];
  float* y = (float*)d_out;

  const size_t MATB = (size_t)NB * NVOX * CCH * sizeof(bf16_t);  // 4 MiB each
  char* ws = (char*)d_ws;
  bf16_t* hT = (bf16_t*)(ws);
  bf16_t* qT = (bf16_t*)(ws + 1 * MATB);
  bf16_t* kT = (bf16_t*)(ws + 2 * MATB);
  bf16_t* vB = (bf16_t*)(ws + 3 * MATB);
  bf16_t* oT = (bf16_t*)(ws + 4 * MATB);
  bf16_t* wb = (bf16_t*)(ws + 5 * MATB);  // 4 x 256*256 bf16 = 512 KiB

  wprep_kernel<<<256, 256, 0, stream>>>(w0, w1, w2, w3, wb);
  gn_kernel<<<NB * 32, 256, 0, stream>>>(x, gamma, beta, hT);
  qkv_kernel<<<2048, 128, 0, stream>>>(hT, wb, b0, b1, b2, qT, kT, vB);
  attn_kernel<<<128, 128, 0, stream>>>(qT, kT, vB, oT);
  out_kernel<<<2048, 128, 0, stream>>>(oT, wb + 3 * 65536, b3, x, y);
}
